// LoRAAttention_68702296867425
// MI455X (gfx1250) — compile-verified, matrix operand order checked by reading, not run on hardware
//
#include <hip/hip_runtime.h>
#include <hip/hip_bf16.h>
#include <math.h>

// ---------------------------------------------------------------------------
// LoRA attention for MI455X (gfx1250, wave32, WMMA + async-to-LDS staging).
// Pipeline:
//   1) W' = W + B@A  (LoRA folded into weights, bf16)      [merge_lora]
//   2) x -> bf16                                            [cvt_f32_bf16]
//   3) Q = x@Wq'^T, K = x@Wk'^T, V^T = (x@Wv'^T)^T          [gemm_nt]
//   4) S = Q@K^T (fp32 out, batched)                        [gemm_nt]
//   5) P = softmax(S/32) -> bf16                            [softmax_rows]
//   6) O = P@V  (= P @ Vt^T, batched)                       [gemm_nt]
//   7) out = O@Ow^T (fp32 out)                              [gemm_nt]
// GEMM: block 128x256, 8 waves, 64x64 wave tiles, K-step 32.
// Operand tiles are staged in LDS via GLOBAL_LOAD_ASYNC_TO_LDS_B128
// (ASYNCcnt), double-buffered, consumed via ds_load_b128 into WMMA frags.
// ---------------------------------------------------------------------------

typedef __bf16 bf16;
typedef bf16  v16bf __attribute__((ext_vector_type(16)));
typedef float v8f   __attribute__((ext_vector_type(8)));

#define E_DIM 1024
#define R_DIM 16
#define BATCH 4
#define SEQ   2048
#define MTOT  (BATCH * SEQ)   // 8192 total rows

// ---------------------- CDNA5 async-copy primitives ------------------------

__device__ __forceinline__ void async_copy_b128(unsigned lds_off, const bf16* gptr) {
  // 16B per lane: memory -> LDS, tracked by ASYNCcnt (no VGPR data path)
  asm volatile("global_load_async_to_lds_b128 %0, %1, off"
               :: "v"(lds_off), "v"(gptr) : "memory");
}

__device__ __forceinline__ void wait_async0() {
  asm volatile("s_wait_asynccnt 0x0" ::: "memory");
}

// ---------------------------- helper kernels -------------------------------

__global__ __launch_bounds__(256) void cvt_f32_bf16(const float* __restrict__ in,
                                                    bf16* __restrict__ out, long n) {
  long i = (long)blockIdx.x * blockDim.x + threadIdx.x;
  if (i < n) out[i] = (bf16)in[i];
}

// out[f][e] = W[f][e] + sum_r Bmat[f][r] * Amat[r][e]   (E x E, bf16 out)
__global__ __launch_bounds__(256) void merge_lora(const float* __restrict__ W,
                                                  const float* __restrict__ Amat,
                                                  const float* __restrict__ Bmat,
                                                  bf16* __restrict__ out) {
  long i = (long)blockIdx.x * blockDim.x + threadIdx.x;
  if (i >= (long)E_DIM * E_DIM) return;
  int f = (int)(i / E_DIM);
  int e = (int)(i % E_DIM);
  float acc = W[i];
#pragma unroll
  for (int r = 0; r < R_DIM; ++r)
    acc += Bmat[f * R_DIM + r] * Amat[r * E_DIM + e];
  out[i] = (bf16)acc;
}

// Row softmax: P[row][c] = softmax(scale * S[row][:])[c], bf16 out.
__global__ __launch_bounds__(256) void softmax_rows(const float* __restrict__ S,
                                                    bf16* __restrict__ P,
                                                    int cols, float scale) {
  long row = blockIdx.x;
  const float* s = S + row * (long)cols;
  bf16* p = P + row * (long)cols;
  __shared__ float red[8];
  int tid = threadIdx.x;

  float m = -3.4e38f;
  for (int c = tid; c < cols; c += 256) m = fmaxf(m, s[c] * scale);
  for (int off = 16; off > 0; off >>= 1) m = fmaxf(m, __shfl_xor(m, off, 32));
  if ((tid & 31) == 0) red[tid >> 5] = m;
  __syncthreads();
  float rowmax = red[0];
#pragma unroll
  for (int i = 1; i < 8; ++i) rowmax = fmaxf(rowmax, red[i]);
  __syncthreads();

  float sum = 0.f;
  for (int c = tid; c < cols; c += 256) sum += __expf(s[c] * scale - rowmax);
  for (int off = 16; off > 0; off >>= 1) sum += __shfl_xor(sum, off, 32);
  if ((tid & 31) == 0) red[tid >> 5] = sum;
  __syncthreads();
  float tot = 0.f;
#pragma unroll
  for (int i = 0; i < 8; ++i) tot += red[i];
  float inv = 1.0f / tot;

  for (int c = tid; c < cols; c += 256)
    p[c] = (bf16)(__expf(s[c] * scale - rowmax) * inv);
}

// ------------------------------- NT GEMM -----------------------------------
// C[m][n] = sum_k A[m][k] * Bm[n][k]   (both operands K-contiguous)
// Block = 256 threads = 8 waves (2x4 grid of 64x64 wave tiles) => 128x256.
// Per K-step(32): async-copy A tile (128x32) + B tile (256x32) into LDS
// (double buffered, 48 KB), then each wave pulls 4 A-frags + 4 B-frags via
// ds_load_b128 and issues 16 v_wmma_f32_16x16x32_bf16.
// TRANS_STORE: scatter C transposed per batch into Vt[b][n][s] (b = m>>11).
// OUT_F32: write fp32 (scores / final output), else bf16.
template <int TRANS_STORE, int OUT_F32>
__global__ __launch_bounds__(256, 1) void gemm_nt(const bf16* __restrict__ A,
                                                  const bf16* __restrict__ Bm,
                                                  void* __restrict__ Cv,
                                                  int N, int K,
                                                  long strA, long strB, long strC) {
  __shared__ bf16 lA[2][128 * 32];   // 8 KB per buffer
  __shared__ bf16 lB[2][256 * 32];   // 16 KB per buffer

  const int tid  = threadIdx.x;
  const int wave = tid >> 5;
  const int lane = tid & 31;
  const int wm = wave & 1;
  const int wn = wave >> 1;
  const int mblk = blockIdx.x * 128;
  const int nblk = blockIdx.y * 256;
  const long zb = blockIdx.z;
  A  += zb * strA;
  Bm += zb * strB;

  const int lr = lane & 15;          // row within fragment
  const int lk = (lane >> 4) << 4;   // K half select (0 or 16)

  v8f acc[4][4];
  const v8f vzero = {0.f, 0.f, 0.f, 0.f, 0.f, 0.f, 0.f, 0.f};
#pragma unroll
  for (int i = 0; i < 4; ++i)
#pragma unroll
    for (int j = 0; j < 4; ++j) acc[i][j] = vzero;

  // Issue async copies for one K-slab into buffer `buf`.
  // A tile: 512 x 16B transfers (2/thread); B tile: 1024 x 16B (4/thread).
  auto issue_tiles = [&](int buf, int k0) {
#pragma unroll
    for (int i = 0; i < 2; ++i) {
      int t = tid + i * 256;
      int row = t >> 2, seg = t & 3;                       // 4 x 16B per row
      const bf16* g = A + (long)(mblk + row) * K + k0 + seg * 8;
      unsigned l = (unsigned)(unsigned long)&lA[buf][row * 32 + seg * 8];
      async_copy_b128(l, g);
    }
#pragma unroll
    for (int i = 0; i < 4; ++i) {
      int t = tid + i * 256;
      int row = t >> 2, seg = t & 3;
      const bf16* g = Bm + (long)(nblk + row) * K + k0 + seg * 8;
      unsigned l = (unsigned)(unsigned long)&lB[buf][row * 32 + seg * 8];
      async_copy_b128(l, g);
    }
  };

  issue_tiles(0, 0);
  const int nsteps = K >> 5;
  for (int ks = 0; ks < nsteps; ++ks) {
    const int buf = ks & 1;
    wait_async0();        // this wave's fills for `buf` are done (ASYNCcnt==0)
    __syncthreads();      // every wave's fills for `buf` are done
    if (ks + 1 < nsteps) issue_tiles(buf ^ 1, (ks + 1) << 5);

    v16bf a[4], b[4];
#pragma unroll
    for (int i = 0; i < 4; ++i)
      a[i] = *(const v16bf*)&lA[buf][(wm * 64 + i * 16 + lr) * 32 + lk];
#pragma unroll
    for (int j = 0; j < 4; ++j)
      b[j] = *(const v16bf*)&lB[buf][(wn * 64 + j * 16 + lr) * 32 + lk];

#pragma unroll
    for (int i = 0; i < 4; ++i)
#pragma unroll
      for (int j = 0; j < 4; ++j)
        acc[i][j] = __builtin_amdgcn_wmma_f32_16x16x32_bf16(
            false, a[i], false, b[j], (short)0, acc[i][j], false, false);
  }

  // C/D layout: VGPR v, lanes 0-15 -> M=v, N=lane; lanes 16-31 -> M=8+v, N=lane-16
  const int cn = lane & 15;
  const int rv = (lane >> 4) << 3;  // 0 or 8
#pragma unroll
  for (int i = 0; i < 4; ++i) {
#pragma unroll
    for (int j = 0; j < 4; ++j) {
#pragma unroll
      for (int v = 0; v < 8; ++v) {
        int r = mblk + wm * 64 + i * 16 + rv + v;
        int c = nblk + wn * 64 + j * 16 + cn;
        float val = acc[i][j][v];
        if (TRANS_STORE) {
          long bb = r >> 11;          // SEQ = 2048 rows per batch
          long ss = r & (SEQ - 1);
          ((bf16*)Cv)[bb * (long)E_DIM * SEQ + (long)c * SEQ + ss] = (bf16)val;
        } else if (OUT_F32) {
          ((float*)Cv)[zb * strC + (long)r * N + c] = val;
        } else {
          ((bf16*)Cv)[zb * strC + (long)r * N + c] = (bf16)val;
        }
      }
    }
  }
}

// ------------------------------- launcher ----------------------------------

extern "C" void kernel_launch(void* const* d_in, const int* in_sizes, int n_in,
                              void* d_out, int out_size, void* d_ws, size_t ws_size,
                              hipStream_t stream) {
  (void)in_sizes; (void)n_in; (void)out_size; (void)ws_size;

  const float* query = (const float*)d_in[0];
  const float* key   = (const float*)d_in[1];
  const float* value = (const float*)d_in[2];
  const float* Qw    = (const float*)d_in[3];
  const float* Kw    = (const float*)d_in[4];
  const float* Vw    = (const float*)d_in[5];
  const float* Qa    = (const float*)d_in[6];
  const float* Qbm   = (const float*)d_in[7];
  const float* Ka    = (const float*)d_in[8];
  const float* Kbm   = (const float*)d_in[9];
  const float* Va    = (const float*)d_in[10];
  const float* Vbm   = (const float*)d_in[11];
  const float* Ow    = (const float*)d_in[12];

  char* ws = (char*)d_ws;
  const size_t MB = 1024 * 1024;
  bf16*  wq = (bf16*)(ws + 0 * MB);     // merged Q weight  (2 MB)
  bf16*  wk = (bf16*)(ws + 2 * MB);
  bf16*  wv = (bf16*)(ws + 4 * MB);
  bf16*  wo = (bf16*)(ws + 6 * MB);
  bf16*  xq = (bf16*)(ws + 8 * MB);     // bf16 inputs (16 MB each)
  bf16*  xk = (bf16*)(ws + 24 * MB);
  bf16*  xv = (bf16*)(ws + 40 * MB);
  bf16*  Qb = (bf16*)(ws + 56 * MB);    // Q projection (16 MB)
  bf16*  Kb = (bf16*)(ws + 72 * MB);    // K projection
  bf16*  Vt = (bf16*)(ws + 88 * MB);    // V projection, transposed per batch
  float* Sf = (float*)(ws + 104 * MB);  // scores fp32 (64 MB)
  bf16*  Pb = (bf16*)(ws + 8 * MB);     // softmax probs (32 MB) — reuses xq/xk
  bf16*  Ob = (bf16*)(ws + 40 * MB);    // attention output (16 MB) — reuses xv

  dim3 blk(256);

  // 1) fold LoRA into weights (bf16), convert O weight
  int wgrid = (E_DIM * E_DIM + 255) / 256;
  merge_lora<<<wgrid, blk, 0, stream>>>(Qw, Qa, Qbm, wq);
  merge_lora<<<wgrid, blk, 0, stream>>>(Kw, Ka, Kbm, wk);
  merge_lora<<<wgrid, blk, 0, stream>>>(Vw, Va, Vbm, wv);
  cvt_f32_bf16<<<wgrid, blk, 0, stream>>>(Ow, wo, (long)E_DIM * E_DIM);

  // 2) inputs -> bf16
  long nx = (long)MTOT * E_DIM;
  int xgrid = (int)((nx + 255) / 256);
  cvt_f32_bf16<<<xgrid, blk, 0, stream>>>(query, xq, nx);
  cvt_f32_bf16<<<xgrid, blk, 0, stream>>>(key,   xk, nx);
  cvt_f32_bf16<<<xgrid, blk, 0, stream>>>(value, xv, nx);

  // 3) QKV projections (M=8192, N=1024, K=1024); V stored transposed
  dim3 gproj(MTOT / 128, E_DIM / 256, 1);
  gemm_nt<0, 0><<<gproj, blk, 0, stream>>>(xq, wq, Qb, E_DIM, E_DIM, 0, 0, 0);
  gemm_nt<0, 0><<<gproj, blk, 0, stream>>>(xk, wk, Kb, E_DIM, E_DIM, 0, 0, 0);
  gemm_nt<1, 0><<<gproj, blk, 0, stream>>>(xv, wv, Vt, E_DIM, E_DIM, 0, 0, 0);

  // 4) scores S = Q @ K^T, batched over B (M=N=2048, K=1024), fp32 out
  dim3 gsc(SEQ / 128, SEQ / 256, BATCH);
  gemm_nt<0, 1><<<gsc, blk, 0, stream>>>(Qb, Kb, Sf, SEQ, E_DIM,
                                         (long)SEQ * E_DIM, (long)SEQ * E_DIM,
                                         (long)SEQ * SEQ);

  // 5) P = softmax(S / sqrt(E))
  softmax_rows<<<BATCH * SEQ, blk, 0, stream>>>(Sf, Pb, SEQ, 0.03125f);

  // 6) O = P @ V  (= P @ Vt^T), batched (M=2048, N=1024, K=2048)
  dim3 gpv(SEQ / 128, E_DIM / 256, BATCH);
  gemm_nt<0, 0><<<gpv, blk, 0, stream>>>(Pb, Vt, Ob, E_DIM, SEQ,
                                         (long)SEQ * SEQ, (long)E_DIM * SEQ,
                                         (long)SEQ * E_DIM);

  // 7) out = O @ Ow^T  (M=8192, N=1024, K=1024), fp32 to d_out
  gemm_nt<0, 1><<<gproj, blk, 0, stream>>>(Ob, wo, d_out, E_DIM, E_DIM, 0, 0, 0);
}
